// KeyedEncoder_30537217475006
// MI455X (gfx1250) — matrix operand order repaired and checked by reading, not compiled, not run on hardware
//
#include <hip/hip_runtime.h>

// ---------------------------------------------------------------------------
// KeyedGRU for MI455X (gfx1250): bf16 WMMA recurrence, fp32 gate math.
// V=50000 E=300 H=1024 B=128 T=512 KEY_LEN=15
// Step kernel: block tile M=64 x N=32 (8 waves, 1 WMMA tile each) to cut
// per-step weight re-streaming from L2 to 2x the weight footprint.
// ---------------------------------------------------------------------------

#define VOCAB   50000
#define EDIM    300
#define EPAD    320       // E padded to multiple of 32 for K-loop
#define HDIM    1024
#define H3      3072
#define BATCH   128
#define TLEN    512
#define KEYLEN  15

typedef __attribute__((ext_vector_type(8)))  float  v8f;
typedef __attribute__((ext_vector_type(8)))  __bf16 v8bf;
typedef __attribute__((ext_vector_type(16))) __bf16 v16bf;

static __device__ inline __bf16 f2bf(float f) {
    union { float f; unsigned u; } in; in.f = f;
    unsigned u = in.u;
    u += 0x7FFFu + ((u >> 16) & 1u);           // round-to-nearest-even
    unsigned short s = (unsigned short)(u >> 16);
    union { unsigned short s; __bf16 b; } out; out.s = s;
    return out.b;
}

static __device__ inline float sigmoidf(float x) { return 1.0f / (1.0f + expf(-x)); }

static __device__ inline v8f wmma_bf16(v16bf a, v16bf b, v8f c) {
    // D = A(16x32 bf16) x B(32x16 bf16) + C(16x16 f32)
    return __builtin_amdgcn_wmma_f32_16x16x32_bf16(
        false, a, false, b, (short)0, c, false, false);
}

static __device__ inline v16bf cat8(v8bf lo, v8bf hi) {
    return __builtin_shufflevector(lo, hi, 0,1,2,3,4,5,6,7,8,9,10,11,12,13,14,15);
}

// --------------------------- conversion kernels ----------------------------

__global__ __launch_bounds__(256)
void cvt_pad_bf16(const float* __restrict__ src, __bf16* __restrict__ dst,
                  int rows, int src_cols, int dst_cols) {
    size_t i = (size_t)blockIdx.x * 256 + threadIdx.x;
    size_t total = (size_t)rows * dst_cols;
    if (i >= total) return;
    int r = (int)(i / dst_cols);
    int c = (int)(i % dst_cols);
    dst[i] = (c < src_cols) ? f2bf(src[(size_t)r * src_cols + c]) : f2bf(0.0f);
}

__global__ __launch_bounds__(256)
void zero_h(float* __restrict__ hf, __bf16* __restrict__ hb, int n) {
    int i = blockIdx.x * 256 + threadIdx.x;
    if (i < n) { hf[i] = 0.0f; hb[i] = f2bf(0.0f); }
}

// ------------------------------ key GRU ------------------------------------
// Batch=1, 15 steps; scalar fp32 (WMMA-hostile shape). One 1024-thread block.

__global__ __launch_bounds__(1024)
void key_gru_kernel(const int* __restrict__ key_ids,
                    const float* __restrict__ emb,
                    const float* __restrict__ W_ih, const float* __restrict__ W_hh,
                    const float* __restrict__ b_ih, const float* __restrict__ b_hh,
                    const float* __restrict__ W_g,  const float* __restrict__ b_g,
                    float* __restrict__ gate_out) {
    __shared__ float h[HDIM];
    __shared__ float xe[EDIM + 4];
    const int g = threadIdx.x;
    h[g] = 0.0f;
    __syncthreads();
    for (int s = 0; s < KEYLEN; ++s) {
        const int tok = key_ids[s];
        if (g < EDIM) xe[g] = emb[(size_t)tok * EDIM + g];
        __syncthreads();
        float xr = b_ih[g], xz = b_ih[HDIM + g], xn = b_ih[2 * HDIM + g];
        const float* wr = W_ih + (size_t)g * EDIM;
        const float* wz = W_ih + (size_t)(HDIM + g) * EDIM;
        const float* wn = W_ih + (size_t)(2 * HDIM + g) * EDIM;
        for (int k = 0; k < EDIM; ++k) {
            float e = xe[k];
            xr += e * wr[k]; xz += e * wz[k]; xn += e * wn[k];
        }
        float hr = b_hh[g], hz = b_hh[HDIM + g], hn = b_hh[2 * HDIM + g];
        const float* vr = W_hh + (size_t)g * HDIM;
        const float* vz = W_hh + (size_t)(HDIM + g) * HDIM;
        const float* vn = W_hh + (size_t)(2 * HDIM + g) * HDIM;
        for (int k = 0; k < HDIM; ++k) {
            float hv = h[k];
            hr += hv * vr[k]; hz += hv * vz[k]; hn += hv * vn[k];
        }
        float r = sigmoidf(xr + hr);
        float z = sigmoidf(xz + hz);
        float n = tanhf(xn + r * hn);
        float hnew = (1.0f - z) * n + z * h[g];
        __syncthreads();
        h[g] = hnew;
        __syncthreads();
    }
    float a = b_g[g];
    const float* wg = W_g + (size_t)g * HDIM;
    for (int k = 0; k < HDIM; ++k) a += h[k] * wg[k];
    gate_out[g] = sigmoidf(a);
}

// --------------------------- main GRU step ---------------------------------
// One launch per timestep. Fused: x-projection GEMM (K=320 from gathered
// bf16 embedding rows) + recurrent GEMM (K=1024 from bf16 h) + gate epilogue.
// Block = 256 threads (8 waves, 4M x 2N). Block tile: M=64 batch rows,
// N=32 h-cols; one 16x16 WMMA tile per wave.
// Grid = (H/32, B/64) = (32, 2): only B/64 = 2 row-blocks re-stream the
// weights each step (16 MB of L2 traffic vs 64 MB with an M=16 tile).

__global__ __launch_bounds__(256)
void gru_step_kernel(const int*    __restrict__ x,        // [B,T]
                     int t,
                     const __bf16* __restrict__ embb,     // [V,EPAD]
                     const __bf16* __restrict__ wihb,     // [3H,EPAD]
                     const __bf16* __restrict__ whhb,     // [3H,H]
                     const float*  __restrict__ b_ih,
                     const float*  __restrict__ b_hh,
                     const float*  __restrict__ gate,     // [H]
                     const float*  __restrict__ hprev_f,  // [B,H]
                     const __bf16* __restrict__ hprev_b,  // [B,H]
                     float*        __restrict__ hnext_f,
                     __bf16*       __restrict__ hnext_b) {
    const int lane = threadIdx.x & 31;
    const int wave = threadIdx.x >> 5;                 // 0..7
    const int m0   = blockIdx.y * 64 + (wave >> 1) * 16; // batch tile base
    const int n0   = blockIdx.x * 32 + (wave & 1) * 16;  // h-column tile base
    const int rA   = lane & 15;                  // A row / B col within tile
    const int kh   = lane >> 4;                  // half selector
    const int kh8  = kh * 8;                     // unused; A K sub-offset is kh16 too
    const int kh16 = kh * 16;                    // B K sub-offset

    v8f acc_r  = {}; // r gate: x-part + h-part combined
    v8f acc_z  = {}; // z gate: combined
    v8f acc_nx = {}; // n gate: x-part only
    v8f acc_nh = {}; // n gate: h-part only (multiplied by r later)

    // ---- phase 1: x-projection, K = EPAD = 320 ----
    const int tok = x[(size_t)(m0 + rA) * TLEN + t];
    const __bf16* arow = embb + (size_t)tok * EPAD;
    const __bf16* bR = wihb + (size_t)(n0 + rA) * EPAD + kh16;
    const __bf16* bZ = wihb + (size_t)(HDIM + n0 + rA) * EPAD + kh16;
    const __bf16* bN = wihb + (size_t)(2 * HDIM + n0 + rA) * EPAD + kh16;
#pragma unroll
    for (int kc = 0; kc < EPAD; kc += 32) {
        v8bf alo = *(const v8bf*)(arow + kc + kh16);
        v8bf ahi = *(const v8bf*)(arow + kc + kh16 + 8);
        v16bf a = cat8(alo, ahi);
        v16bf fr = *(const v16bf*)(bR + kc);
        v16bf fz = *(const v16bf*)(bZ + kc);
        v16bf fn = *(const v16bf*)(bN + kc);
        acc_r  = wmma_bf16(a, fr, acc_r);
        acc_z  = wmma_bf16(a, fz, acc_z);
        acc_nx = wmma_bf16(a, fn, acc_nx);
    }

    // ---- phase 2: recurrence, K = H = 1024 ----
    const __bf16* hrow = hprev_b + (size_t)(m0 + rA) * HDIM;
    const __bf16* cR = whhb + (size_t)(n0 + rA) * HDIM + kh16;
    const __bf16* cZ = whhb + (size_t)(HDIM + n0 + rA) * HDIM + kh16;
    const __bf16* cN = whhb + (size_t)(2 * HDIM + n0 + rA) * HDIM + kh16;
#pragma unroll 4
    for (int kc = 0; kc < HDIM; kc += 32) {
        v8bf alo = *(const v8bf*)(hrow + kc + kh16);
        v8bf ahi = *(const v8bf*)(hrow + kc + kh16 + 8);
        v16bf a = cat8(alo, ahi);
        v16bf fr = *(const v16bf*)(cR + kc);
        v16bf fz = *(const v16bf*)(cZ + kc);
        v16bf fn = *(const v16bf*)(cN + kc);
        acc_r  = wmma_bf16(a, fr, acc_r);
        acc_z  = wmma_bf16(a, fz, acc_z);
        acc_nh = wmma_bf16(a, fn, acc_nh);
    }

    // ---- epilogue: gates + keyed gating ----
    // C/D layout: vgpr i -> (M = i + 8*kh, N = lane&15)
    const int hcol = n0 + (lane & 15);
    const float brz0 = b_ih[hcol]            + b_hh[hcol];
    const float brz1 = b_ih[HDIM + hcol]     + b_hh[HDIM + hcol];
    const float bnx  = b_ih[2 * HDIM + hcol];
    const float bnh  = b_hh[2 * HDIM + hcol];
    const float gk   = gate[hcol];
#pragma unroll
    for (int i = 0; i < 8; ++i) {
        const int b = m0 + kh * 8 + i;
        const float r = sigmoidf(acc_r[i] + brz0);
        const float z = sigmoidf(acc_z[i] + brz1);
        const float n = tanhf(acc_nx[i] + bnx + r * (acc_nh[i] + bnh));
        const float hp = hprev_f[(size_t)b * HDIM + hcol];
        float hn = (1.0f - z) * n + z * hp;
        hn *= gk;                                   // key-conditioned gate
        hnext_f[(size_t)b * HDIM + hcol] = hn;
        hnext_b[(size_t)b * HDIM + hcol] = f2bf(hn);
    }
}

// ------------------------------ launcher -----------------------------------

extern "C" void kernel_launch(void* const* d_in, const int* in_sizes, int n_in,
                              void* d_out, int out_size, void* d_ws, size_t ws_size,
                              hipStream_t stream) {
    const int*   x       = (const int*)  d_in[0];   // [B,T]
    const int*   key_ids = (const int*)  d_in[1];   // [1,KEY_LEN]
    const float* emb     = (const float*)d_in[2];   // [V,E]
    const float* W_ih    = (const float*)d_in[3];   // [3H,E]
    const float* W_hh    = (const float*)d_in[4];   // [3H,H]
    const float* b_ih    = (const float*)d_in[5];   // [3H]
    const float* b_hh    = (const float*)d_in[6];   // [3H]
    const float* W_g     = (const float*)d_in[7];   // [H,H]
    const float* b_g     = (const float*)d_in[8];   // [H]
    float* out = (float*)d_out;                     // [B,H]

    // workspace carve-up (256B aligned)
    char* ws = (char*)d_ws;
    size_t off = 0;
    auto carve = [&](size_t bytes) {
        char* p = ws + off;
        off += (bytes + 255) & ~(size_t)255;
        return p;
    };
    __bf16* embb = (__bf16*)carve((size_t)VOCAB * EPAD * 2);   // 32.0 MB
    __bf16* wihb = (__bf16*)carve((size_t)H3 * EPAD * 2);      //  2.0 MB
    __bf16* whhb = (__bf16*)carve((size_t)H3 * HDIM * 2);      //  6.3 MB
    float*  gate = (float*) carve((size_t)HDIM * 4);
    float*  hf0  = (float*) carve((size_t)BATCH * HDIM * 4);
    float*  hf1  = (float*) carve((size_t)BATCH * HDIM * 4);
    __bf16* hb0  = (__bf16*)carve((size_t)BATCH * HDIM * 2);
    __bf16* hb1  = (__bf16*)carve((size_t)BATCH * HDIM * 2);
    (void)ws_size; (void)in_sizes; (void)n_in; (void)out_size;

    // 1) convert/pad weights + embedding table to bf16
    {
        size_t n = (size_t)VOCAB * EPAD;
        cvt_pad_bf16<<<dim3((unsigned)((n + 255) / 256)), 256, 0, stream>>>(
            emb, embb, VOCAB, EDIM, EPAD);
    }
    {
        size_t n = (size_t)H3 * EPAD;
        cvt_pad_bf16<<<dim3((unsigned)((n + 255) / 256)), 256, 0, stream>>>(
            W_ih, wihb, H3, EDIM, EPAD);
    }
    {
        size_t n = (size_t)H3 * HDIM;
        cvt_pad_bf16<<<dim3((unsigned)((n + 255) / 256)), 256, 0, stream>>>(
            W_hh, whhb, H3, HDIM, HDIM);
    }

    // 2) h0 = 0
    {
        int n = BATCH * HDIM;
        zero_h<<<dim3((n + 255) / 256), 256, 0, stream>>>(hf0, hb0, n);
    }

    // 3) key GRU -> gate
    key_gru_kernel<<<1, 1024, 0, stream>>>(key_ids, emb, W_ih, W_hh,
                                           b_ih, b_hh, W_g, b_g, gate);

    // 4) main keyed recurrence: 512 sequential WMMA step kernels (ping-pong h)
    const dim3 grid(HDIM / 32, BATCH / 64);   // (32, 2)
    for (int t = 0; t < TLEN; ++t) {
        const float*  hf_in  = (t & 1) ? hf1 : hf0;
        float*        hf_out = (t & 1) ? hf0 : hf1;
        const __bf16* hb_in  = (t & 1) ? hb1 : hb0;
        __bf16*       hb_out = (t & 1) ? hb0 : hb1;
        gru_step_kernel<<<grid, 256, 0, stream>>>(
            x, t, embb, wihb, whhb, b_ih, b_hh, gate,
            hf_in, hb_in, hf_out, hb_out);
    }

    // 5) final hidden (t=511 odd -> wrote hf0) -> d_out
    hipMemcpyAsync(out, hf0, (size_t)BATCH * HDIM * sizeof(float),
                   hipMemcpyDeviceToDevice, stream);
}
